// Attention_Feedback_GRU_80092550136358
// MI455X (gfx1250) — matrix operand order repaired and checked by reading, not compiled, not run on hardware
//
#include <hip/hip_runtime.h>

#define B_   64
#define T_   64
#define D_   256
#define TA_  128
#define U_   256
#define V_   50257
#define VP_  50272      // V padded to multiple of 32
#define K3_  768        // 3*U
#define SPLITS 64

typedef __attribute__((ext_vector_type(16))) __bf16 v16bf;
typedef __attribute__((ext_vector_type(8)))  float  v8f;

union Frag { v16bf v; uint4 u[2]; };

__device__ __forceinline__ unsigned short f2bf(float f) {
    unsigned int u = __float_as_uint(f);
    unsigned int r = (u + 0x7FFFu + ((u >> 16) & 1u)) >> 16;
    return (unsigned short)r;
}
__device__ __forceinline__ float hsig(float x) {
    return fminf(fmaxf(0.2f * x + 0.5f, 0.0f), 1.0f);
}
#define WMMA_BF16(A, Bf, C) __builtin_amdgcn_wmma_f32_16x16x32_bf16(false, (A).v, false, (Bf).v, (short)0, (C), false, false)

// ---------------- one-time conversion kernels ----------------
__global__ void k_convert_f32_bf16(const float* __restrict__ src, unsigned short* __restrict__ dst, int n) {
    int i = blockIdx.x * 256 + threadIdx.x;
    if (i < n) dst[i] = f2bf(src[i]);
}
// kernel (D x 3U) -> kT (3U x D) bf16
__global__ void k_convert_kT(const float* __restrict__ K, unsigned short* __restrict__ kT) {
    int i = blockIdx.x * 256 + threadIdx.x;   // over 768*256
    int n = i >> 8;                            // 0..767
    int d = i & 255;
    kT[i] = f2bf(K[d * K3_ + n]);
}
// W_o1 (U x V) -> WT (VP x U) bf16, zero-padded rows
__global__ void k_convert_wo1T(const float* __restrict__ W, unsigned short* __restrict__ WT) {
    int i = blockIdx.x * 256 + threadIdx.x;    // over VP_*256 (fits int)
    int v = i >> 8;
    int u = i & 255;
    WT[i] = (v < V_) ? f2bf(W[u * V_ + v]) : (unsigned short)0;
}
// embedding (V x D) -> ET (D x VP) bf16, zero-padded cols
__global__ void k_convert_embT(const float* __restrict__ E, unsigned short* __restrict__ ET) {
    int i = blockIdx.x * 256 + threadIdx.x;    // over 256*VP_
    int n = i / VP_;
    int k = i - n * VP_;
    ET[i] = (k < V_) ? f2bf(E[k * D_ + n]) : (unsigned short)0;
}
__global__ void k_zero(float* __restrict__ p, int n) {
    int i = blockIdx.x * 256 + threadIdx.x;
    if (i < n) p[i] = 0.0f;
}

// ---------------- xp = x @ kernel + bias  (4096 x 256 x 768), M-blocked x4 ----------------
__global__ void k_gemm_xp(const unsigned short* __restrict__ Xbf, const unsigned short* __restrict__ kT,
                          const float* __restrict__ bias, float* __restrict__ xp) {
    int lane = threadIdx.x & 31, wave = threadIdx.x >> 5;
    int gid = blockIdx.x * 8 + wave;           // 0..3071
    int nt = gid % 48;
    int mg = gid / 48;                         // 0..63 (64-row M group)
    int m0 = mg * 64, n0 = nt * 16;
    int am   = lane & 15;
    int bcol = n0 + am;
    int ahi  = (lane >> 4) * 8;
    int bhi  = (lane >> 4) * 8;
    v8f acc0 = {}, acc1 = {}, acc2 = {}, acc3 = {};
    #pragma unroll
    for (int k = 0; k < 256; k += 32) {
        Frag b, a0, a1, a2, a3;
        const uint4* pb = (const uint4*)(kT + bcol * 256 + k + bhi);
        b.u[0] = pb[0]; b.u[1] = pb[2];
        const unsigned short* ap = Xbf + (m0 + am) * 256 + k;
        a0.u[0] = *(const uint4*)(ap + ahi);            a0.u[1] = *(const uint4*)(ap + 16 + ahi);
        a1.u[0] = *(const uint4*)(ap + 16 * 256 + ahi); a1.u[1] = *(const uint4*)(ap + 16 * 256 + 16 + ahi);
        a2.u[0] = *(const uint4*)(ap + 32 * 256 + ahi); a2.u[1] = *(const uint4*)(ap + 32 * 256 + 16 + ahi);
        a3.u[0] = *(const uint4*)(ap + 48 * 256 + ahi); a3.u[1] = *(const uint4*)(ap + 48 * 256 + 16 + ahi);
        acc0 = WMMA_BF16(a0, b, acc0);
        acc1 = WMMA_BF16(a1, b, acc1);
        acc2 = WMMA_BF16(a2, b, acc2);
        acc3 = WMMA_BF16(a3, b, acc3);
    }
    int n  = n0 + am;
    int mb = m0 + (lane >> 4) * 8;
    float bv = bias[n];
    #pragma unroll
    for (int i = 0; i < 8; ++i) {
        xp[(mb + i) * K3_ + n]      = acc0[i] + bv;
        xp[(mb + 16 + i) * K3_ + n] = acc1[i] + bv;
        xp[(mb + 32 + i) * K3_ + n] = acc2[i] + bv;
        xp[(mb + 48 + i) * K3_ + n] = acc3[i] + bv;
    }
}

// ---------------- fused attention + two tanh layers, per batch element ----------------
__global__ void k_attn_s(const float* __restrict__ att, const float* __restrict__ h,
                         const float* __restrict__ aak, const float* __restrict__ aab,
                         unsigned short* __restrict__ s2bf) {
    __shared__ float sh_h[U_], sh_e[TA_], sh_c[U_], sh_s1[U_], sh_inv;
    int b = blockIdx.x, tid = threadIdx.x;
    sh_h[tid] = h[b * U_ + tid];
    __syncthreads();
    if (tid < TA_) {
        const float* ar = att + ((long)b * TA_ + tid) * U_;
        float acc = 0.f;
        for (int u = 0; u < U_; ++u) acc += ar[u] * sh_h[u];
        sh_e[tid] = acc;
    }
    __syncthreads();
    if (tid == 0) {
        float m = sh_e[0];
        for (int t = 1; t < TA_; ++t) m = fmaxf(m, sh_e[t]);
        float s = 0.f;
        for (int t = 0; t < TA_; ++t) { float ev = __expf(sh_e[t] - m); sh_e[t] = ev; s += ev; }
        sh_inv = 1.0f / s;
    }
    __syncthreads();
    float inv = sh_inv;
    {   // context vector c = e . att
        float acc = 0.f;
        for (int t = 0; t < TA_; ++t) acc += sh_e[t] * att[((long)b * TA_ + t) * U_ + tid];
        sh_c[tid] = acc * inv;
    }
    __syncthreads();
    {   // s1 = tanh(c @ Wa0 + ba0)
        float acc = aab[tid];
        for (int d = 0; d < U_; ++d) acc += sh_c[d] * aak[d * 512 + tid];
        sh_s1[tid] = tanhf(acc);
    }
    __syncthreads();
    {   // s2 = tanh(s1 @ Wa1 + ba1), emit bf16 for the vocab GEMM
        float acc = aab[256 + tid];
        for (int d = 0; d < U_; ++d) acc += sh_s1[d] * aak[d * 512 + 256 + tid];
        s2bf[b * U_ + tid] = f2bf(tanhf(acc));
    }
}

// ---------------- logits = s2 @ W_o1  (64 x 256 x 50272), one wave = 64x16 strip ----------------
__global__ void k_gemm_logits(const unsigned short* __restrict__ Abf, const unsigned short* __restrict__ WT,
                              float* __restrict__ L) {
    int lane = threadIdx.x & 31, wave = threadIdx.x >> 5;
    int nt = blockIdx.x * 8 + wave;            // N tile
    if (nt >= VP_ / 16) return;
    int n0 = nt * 16;
    int am   = lane & 15;
    int bcol = n0 + am;
    int ahi  = (lane >> 4) * 8;
    int bhi  = (lane >> 4) * 8;
    v8f acc0 = {}, acc1 = {}, acc2 = {}, acc3 = {};
    #pragma unroll
    for (int k = 0; k < 256; k += 32) {
        Frag b, a0, a1, a2, a3;
        const uint4* pb = (const uint4*)(WT + (long)bcol * 256 + k + bhi);
        b.u[0] = pb[0]; b.u[1] = pb[2];
        const unsigned short* ap = Abf + am * 256 + k;
        a0.u[0] = *(const uint4*)(ap + ahi);            a0.u[1] = *(const uint4*)(ap + 16 + ahi);
        a1.u[0] = *(const uint4*)(ap + 16 * 256 + ahi); a1.u[1] = *(const uint4*)(ap + 16 * 256 + 16 + ahi);
        a2.u[0] = *(const uint4*)(ap + 32 * 256 + ahi); a2.u[1] = *(const uint4*)(ap + 32 * 256 + 16 + ahi);
        a3.u[0] = *(const uint4*)(ap + 48 * 256 + ahi); a3.u[1] = *(const uint4*)(ap + 48 * 256 + 16 + ahi);
        acc0 = WMMA_BF16(a0, b, acc0);
        acc1 = WMMA_BF16(a1, b, acc1);
        acc2 = WMMA_BF16(a2, b, acc2);
        acc3 = WMMA_BF16(a3, b, acc3);
    }
    int n  = n0 + am;
    int mb = (lane >> 4) * 8;
    #pragma unroll
    for (int i = 0; i < 8; ++i) {
        L[(long)(mb + i) * VP_ + n]      = acc0[i];
        L[(long)(mb + 16 + i) * VP_ + n] = acc1[i];
        L[(long)(mb + 32 + i) * VP_ + n] = acc2[i];
        L[(long)(mb + 48 + i) * VP_ + n] = acc3[i];
    }
}

// ---------------- row softmax over V, emit bf16 p (zero-padded to VP) ----------------
__global__ void k_softmax(const float* __restrict__ L, unsigned short* __restrict__ P) {
    __shared__ float red[256];
    int r = blockIdx.x, tid = threadIdx.x;
    const float* row = L + (long)r * VP_;
    float m = -3.4e38f;
    for (int v = tid; v < V_; v += 256) m = fmaxf(m, row[v]);
    red[tid] = m; __syncthreads();
    for (int s = 128; s > 0; s >>= 1) { if (tid < s) red[tid] = fmaxf(red[tid], red[tid + s]); __syncthreads(); }
    m = red[0]; __syncthreads();
    float sum = 0.f;
    for (int v = tid; v < V_; v += 256) sum += __expf(row[v] - m);
    red[tid] = sum; __syncthreads();
    for (int s = 128; s > 0; s >>= 1) { if (tid < s) red[tid] += red[tid + s]; __syncthreads(); }
    float inv = 1.0f / red[0];
    unsigned short* prow = P + (long)r * VP_;
    for (int v = tid; v < V_; v += 256) prow[v] = f2bf(__expf(row[v] - m) * inv);
    for (int v = V_ + tid; v < VP_; v += 256) prow[v] = 0;
}

// ---------------- fb = p @ embedding  (64 x 50272 x 256) — deterministic split-K, M-blocked x4 ----------------
__global__ void k_gemm_fb(const unsigned short* __restrict__ P, const unsigned short* __restrict__ ET,
                          float* __restrict__ fbpart) {
    int lane = threadIdx.x & 31, wave = threadIdx.x >> 5;
    int gid = blockIdx.x * 8 + wave;           // 0..1023
    int split = gid & (SPLITS - 1);
    int nt    = gid >> 6;                      // 0..15
    int n0 = nt * 16;
    int am   = lane & 15;
    int brow = n0 + am;
    int ahi  = (lane >> 4) * 8;
    int bhi  = (lane >> 4) * 8;
    v8f acc0 = {}, acc1 = {}, acc2 = {}, acc3 = {};
    const int nk = VP_ / 32;                   // 1571
    for (int ks = split; ks < nk; ks += SPLITS) {
        int k = ks * 32;
        Frag b, a0, a1, a2, a3;
        const uint4* pb = (const uint4*)(ET + (long)brow * VP_ + k + bhi);
        __builtin_prefetch(ET + (long)brow * VP_ + k + SPLITS * 32, 0, 0);
        b.u[0] = pb[0]; b.u[1] = pb[2];
        const unsigned short* ap = P + (long)am * VP_ + k;
        a0.u[0] = *(const uint4*)(ap + ahi);                  a0.u[1] = *(const uint4*)(ap + 16 + ahi);
        a1.u[0] = *(const uint4*)(ap + (long)16 * VP_ + ahi); a1.u[1] = *(const uint4*)(ap + (long)16 * VP_ + 16 + ahi);
        a2.u[0] = *(const uint4*)(ap + (long)32 * VP_ + ahi); a2.u[1] = *(const uint4*)(ap + (long)32 * VP_ + 16 + ahi);
        a3.u[0] = *(const uint4*)(ap + (long)48 * VP_ + ahi); a3.u[1] = *(const uint4*)(ap + (long)48 * VP_ + 16 + ahi);
        acc0 = WMMA_BF16(a0, b, acc0);
        acc1 = WMMA_BF16(a1, b, acc1);
        acc2 = WMMA_BF16(a2, b, acc2);
        acc3 = WMMA_BF16(a3, b, acc3);
    }
    int n  = n0 + am;
    int mb = (lane >> 4) * 8;
    float* fp = fbpart + split * (B_ * U_);
    #pragma unroll
    for (int i = 0; i < 8; ++i) {
        fp[(mb + i) * U_ + n]      = acc0[i];
        fp[(mb + 16 + i) * U_ + n] = acc1[i];
        fp[(mb + 32 + i) * U_ + n] = acc2[i];
        fp[(mb + 48 + i) * U_ + n] = acc3[i];
    }
}
__global__ void k_reduce_fb(const float* __restrict__ fbpart, float* __restrict__ fb) {
    int i = blockIdx.x * 256 + threadIdx.x;    // over 64*256
    float acc = 0.f;
    for (int s = 0; s < SPLITS; ++s) acc += fbpart[s * (B_ * U_) + i];
    fb[i] = acc;
}

// ---------------- GRU gates, per batch element ----------------
__global__ void k_gru(const float* __restrict__ xp, const float* __restrict__ fb,
                      const float* __restrict__ K, const float* __restrict__ RK,
                      float* __restrict__ h, float* __restrict__ out, int t) {
    __shared__ float sh_fb[U_], sh_h[U_], sh_rh[U_];
    int b = blockIdx.x, j = threadIdx.x;
    sh_fb[j] = fb[b * U_ + j];
    sh_h[j]  = h[b * U_ + j];
    __syncthreads();
    float az = 0.f, ar = 0.f, ah = 0.f, hz = 0.f, hr = 0.f;
    for (int d = 0; d < U_; ++d) {
        float fv = sh_fb[d], hv = sh_h[d];
        const float* kr = K  + d * K3_;
        const float* rr = RK + d * K3_;
        az += fv * kr[j];        ar += fv * kr[256 + j];  ah += fv * kr[512 + j];
        hz += hv * rr[j];        hr += hv * rr[256 + j];
    }
    const float* xpr = xp + ((long)(b * T_ + t)) * K3_;
    float z = hsig(xpr[j] + az + hz);
    float r = hsig(xpr[256 + j] + ar + hr);
    sh_rh[j] = r * sh_h[j];
    __syncthreads();
    float hh = 0.f;
    for (int d = 0; d < U_; ++d) hh += sh_rh[d] * RK[d * K3_ + 512 + j];
    hh = tanhf(xpr[512 + j] + ah + hh);
    float hn = z * sh_h[j] + (1.0f - z) * hh;
    h[b * U_ + j] = hn;
    out[((long)b * T_ + t) * U_ + j] = hn;
}

extern "C" void kernel_launch(void* const* d_in, const int* in_sizes, int n_in,
                              void* d_out, int out_size, void* d_ws, size_t ws_size,
                              hipStream_t stream) {
    const float* x    = (const float*)d_in[0];
    const float* att  = (const float*)d_in[1];
    const float* Km   = (const float*)d_in[2];
    const float* RK   = (const float*)d_in[3];
    const float* bias = (const float*)d_in[4];
    const float* aak  = (const float*)d_in[5];
    const float* aab  = (const float*)d_in[6];
    const float* Wo1  = (const float*)d_in[7];
    const float* emb  = (const float*)d_in[8];
    float* out = (float*)d_out;

    char* ws = (char*)d_ws;
    size_t off = 0;
    auto alloc = [&](size_t bytes) -> void* {
        void* p = ws + off;
        off += (bytes + 255) & ~(size_t)255;
        return p;
    };
    unsigned short* xbf    = (unsigned short*)alloc((size_t)B_ * T_ * D_ * 2);   // 2 MB
    unsigned short* kT     = (unsigned short*)alloc((size_t)K3_ * D_ * 2);       // 0.4 MB
    unsigned short* WT     = (unsigned short*)alloc((size_t)VP_ * U_ * 2);       // 25.7 MB
    unsigned short* ET     = (unsigned short*)alloc((size_t)D_ * VP_ * 2);       // 25.7 MB
    float*          xp     = (float*)alloc((size_t)B_ * T_ * K3_ * 4);           // 12.6 MB
    float*          L      = (float*)alloc((size_t)B_ * VP_ * 4);                // 12.9 MB
    unsigned short* P      = (unsigned short*)alloc((size_t)B_ * VP_ * 2);       // 6.4 MB
    float*          fbpart = (float*)alloc((size_t)SPLITS * B_ * U_ * 4);        // 4 MB
    float*          hbuf   = (float*)alloc((size_t)B_ * U_ * 4);
    unsigned short* s2bf   = (unsigned short*)alloc((size_t)B_ * U_ * 2);
    float*          fb     = (float*)alloc((size_t)B_ * U_ * 4);

    dim3 blk(256);
    k_convert_f32_bf16<<<(B_ * T_ * D_) / 256, blk, 0, stream>>>(x, xbf, B_ * T_ * D_);
    k_convert_kT<<<(K3_ * D_) / 256, blk, 0, stream>>>(Km, kT);
    k_convert_wo1T<<<(VP_ * U_) / 256, blk, 0, stream>>>(Wo1, WT);
    k_convert_embT<<<(D_ * VP_) / 256, blk, 0, stream>>>(emb, ET);
    k_zero<<<(B_ * U_) / 256, blk, 0, stream>>>(hbuf, B_ * U_);
    k_gemm_xp<<<(64 * 48) / 8, blk, 0, stream>>>(xbf, kT, bias, xp);

    const int logit_blocks = ((VP_ / 16) + 7) / 8;   // 393
    for (int t = 0; t < T_; ++t) {
        k_attn_s<<<B_, blk, 0, stream>>>(att, hbuf, aak, aab, s2bf);
        k_gemm_logits<<<logit_blocks, blk, 0, stream>>>(s2bf, WT, L);
        k_softmax<<<B_, blk, 0, stream>>>(L, P);
        k_gemm_fb<<<(16 * SPLITS) / 8, blk, 0, stream>>>(P, ET, fbpart);
        k_reduce_fb<<<(B_ * U_) / 256, blk, 0, stream>>>(fbpart, fb);
        k_gru<<<B_, blk, 0, stream>>>(xp, fb, Km, RK, hbuf, out, t);
    }
}
